// NeuronCrossAttention_18270790877581
// MI455X (gfx1250) — compile-verified, matrix operand order checked by reading, not run on hardware
//
#include <hip/hip_runtime.h>
#include <hip/hip_bf16.h>

// ---- problem constants ----
#define BB 4
#define SS 512
#define KVL 4096
#define DM 1024
#define HH 16
#define HD 64

typedef __attribute__((ext_vector_type(16))) __bf16 bf16x16;
typedef __attribute__((ext_vector_type(8)))  __bf16 bf16x8;
typedef __attribute__((ext_vector_type(4)))  __bf16 bf16x4;
typedef __attribute__((ext_vector_type(8)))  float  f32x8;

// Load one 16-bit WMMA fragment (A or B role) for wave32.
// Lane l (l<16): elements k+{0..7, 16..23}; lane l+16: k+{8..15, 24..31}.
__device__ __forceinline__ bf16x16 load_frag(const __bf16* p, int hv) {
    union { bf16x16 v; bf16x8 h[2]; } u;
    u.h[0] = *reinterpret_cast<const bf16x8*>(p + hv * 8);
    u.h[1] = *reinterpret_cast<const bf16x8*>(p + 16 + hv * 8);
    return u.v;
}

__device__ __forceinline__ f32x8 wmma_bf16(bf16x16 a, bf16x16 b, f32x8 c) {
    return __builtin_amdgcn_wmma_f32_16x16x32_bf16(
        false, a, false, b, (short)0, c, false, false);
}

// ---- f32 -> bf16 conversion (vectorized x4) ----
__global__ __launch_bounds__(256) void cvt_kernel(const float* __restrict__ src,
                                                  __bf16* __restrict__ dst, int n) {
    int i = (blockIdx.x * 256 + threadIdx.x) * 4;
    if (i + 3 < n) {
        float4 f = *reinterpret_cast<const float4*>(src + i);
        bf16x4 o = { (__bf16)f.x, (__bf16)f.y, (__bf16)f.z, (__bf16)f.w };
        *reinterpret_cast<bf16x4*>(dst + i) = o;
    }
}

// ---- generic WMMA projection: Y[rows, D] = X[rows, D] @ W^T (+ bias) ----
// MODE 0: Q  -> outB = qb  in [B,H,S,HD] bf16
// MODE 1: K  -> outF = cache_k [B,H,KV,HD] f32, outB = same layout bf16
// MODE 2: V  -> outF = cache_v [B,H,KV,HD] f32, outB = vT [B,H,HD,KV] bf16
// MODE 3: O  -> outF = out [rows, D] f32
template <int MODE>
__global__ __launch_bounds__(256) void proj_kernel(
    const __bf16* __restrict__ X, const __bf16* __restrict__ W,
    const float* __restrict__ bias, float* __restrict__ outF,
    __bf16* __restrict__ outB, int rows)
{
    int wave = (int)((blockIdx.x * blockDim.x + threadIdx.x) >> 5);
    int lane = threadIdx.x & 31;
    int mt = wave >> 4;            // D/64 = 16 N-tiles per M-tile
    int nt = wave & 15;
    if (mt >= (rows >> 4)) return;
    int m0 = mt << 4, n0 = nt << 6;
    int hv = lane >> 4, l15 = lane & 15;

    const __bf16* ap  = X + (size_t)(m0 + l15) * DM;
    const __bf16* bp0 = W + (size_t)(n0 +  0 + l15) * DM;
    const __bf16* bp1 = W + (size_t)(n0 + 16 + l15) * DM;
    const __bf16* bp2 = W + (size_t)(n0 + 32 + l15) * DM;
    const __bf16* bp3 = W + (size_t)(n0 + 48 + l15) * DM;

    f32x8 c0 = {}, c1 = {}, c2 = {}, c3 = {};
    for (int k = 0; k < DM; k += 32) {
        bf16x16 a = load_frag(ap + k, hv);
        c0 = wmma_bf16(a, load_frag(bp0 + k, hv), c0);
        c1 = wmma_bf16(a, load_frag(bp1 + k, hv), c1);
        c2 = wmma_bf16(a, load_frag(bp2 + k, hv), c2);
        c3 = wmma_bf16(a, load_frag(bp3 + k, hv), c3);
    }
    f32x8 cs[4] = {c0, c1, c2, c3};
    #pragma unroll
    for (int t = 0; t < 4; ++t) {
        int o = n0 + t * 16 + l15;
        float bb = 0.f;
        if (MODE == 0 || MODE == 2 || MODE == 3) bb = bias[o];
        #pragma unroll
        for (int v = 0; v < 8; ++v) {
            int m = m0 + v + 8 * hv;           // global row
            float y = cs[t][v] + bb;
            if (MODE == 0) {                   // q -> [B,H,S,HD]
                int b = m >> 9, s = m & (SS - 1), h = o >> 6, hd = o & 63;
                outB[(((size_t)(b * HH + h) * SS) + s) * HD + hd] = (__bf16)y;
            } else if (MODE == 1) {            // k
                int b = m >> 12, kv = m & (KVL - 1), h = o >> 6, hd = o & 63;
                size_t idx = ((size_t)(b * HH + h) * KVL + kv) * HD + hd;
                outF[idx] = y;
                outB[idx] = (__bf16)y;
            } else if (MODE == 2) {            // v + transposed bf16
                int b = m >> 12, kv = m & (KVL - 1), h = o >> 6, hd = o & 63;
                size_t idxF = ((size_t)(b * HH + h) * KVL + kv) * HD + hd;
                outF[idxF] = y;
                size_t idxT = ((size_t)(b * HH + h) * HD + hd) * KVL + kv;
                outB[idxT] = (__bf16)y;
            } else {                           // final out
                outF[(size_t)m * DM + o] = y;
            }
        }
    }
}

// ---- flash attention: one wave per (b,h, 16-query tile) ----
// Row sums of the prob matrix are produced by an extra WMMA against an
// all-ones B fragment (osum), so only the row-max needs lane shuffles.
__global__ __launch_bounds__(128) void attn_kernel(
    const __bf16* __restrict__ qb, const __bf16* __restrict__ kb,
    const __bf16* __restrict__ vT, __bf16* __restrict__ attnb)
{
    const float LOG2E = 1.4426950408889634f;
    int wid  = threadIdx.x >> 5;
    int lane = threadIdx.x & 31;
    int gw = blockIdx.x * 4 + wid;     // 0 .. B*H*(S/16)-1 = 2047
    int qt = gw & 31;                  // S/16 = 32 query tiles
    int bh = gw >> 5;                  // 0..63  (b*H + h)
    int hv = lane >> 4, l15 = lane & 15;

    __shared__ __align__(16) __bf16 plds[4][16 * 32];
    __bf16* pl = plds[wid];

    const __bf16* qp = qb + ((size_t)bh * SS + qt * 16 + l15) * HD;
    bf16x16 qa0 = load_frag(qp, hv);
    bf16x16 qa1 = load_frag(qp + 32, hv);
    const __bf16* kbase = kb + (size_t)bh * KVL * HD;
    const __bf16* vbase = vT + (size_t)bh * HD * KVL;

    bf16x16 ones;
    #pragma unroll
    for (int i = 0; i < 16; ++i) ones[i] = (__bf16)1.0f;

    float rm[8];
    f32x8 o0 = {}, o1 = {}, o2 = {}, o3 = {}, osum = {};
    #pragma unroll
    for (int v = 0; v < 8; ++v) rm[v] = -1e30f;

    for (int j = 0; j < KVL / 32; ++j) {
        int kv0 = j * 32;
        f32x8 z = {};
        const __bf16* kp0 = kbase + (size_t)(kv0 + l15) * HD;
        f32x8 s0 = wmma_bf16(qa0, load_frag(kp0, hv), z);
        s0 = wmma_bf16(qa1, load_frag(kp0 + 32, hv), s0);
        const __bf16* kp1 = kbase + (size_t)(kv0 + 16 + l15) * HD;
        f32x8 s1 = wmma_bf16(qa0, load_frag(kp1, hv), z);
        s1 = wmma_bf16(qa1, load_frag(kp1 + 32, hv), s1);
        if (j < KVL / 32 - 1) {
            __builtin_prefetch(kbase + (size_t)(kv0 + 32 + l15) * HD, 0, 1);
            __builtin_prefetch(vbase + (size_t)l15 * KVL + kv0 + 32, 0, 1);
        }

        // online softmax over the 32-wide chunk (rows live in 16-lane halves)
        #pragma unroll
        for (int v = 0; v < 8; ++v) {
            float a = s0[v] * 0.125f;          // 1/sqrt(64)
            float b = s1[v] * 0.125f;
            float mx = fmaxf(a, b);
            #pragma unroll
            for (int ofs = 1; ofs < 16; ofs <<= 1)
                mx = fmaxf(mx, __shfl_xor(mx, ofs, 32));
            float nm = fmaxf(rm[v], mx);
            float f  = exp2f((rm[v] - nm) * LOG2E);
            float p0 = exp2f((a - nm) * LOG2E);
            float p1 = exp2f((b - nm) * LOG2E);
            rm[v] = nm;
            o0[v] *= f; o1[v] *= f; o2[v] *= f; o3[v] *= f; osum[v] *= f;
            int r = (v + 8 * hv) * 32;
            pl[r + l15]      = (__bf16)p0;     // C-layout -> LDS (row-major 16x32)
            pl[r + 16 + l15] = (__bf16)p1;
        }
        // re-load probs in A-fragment layout; P @ V plus row-sum via ones-GEMM
        bf16x16 pa = load_frag(pl + l15 * 32, hv);
        osum = wmma_bf16(pa, ones, osum);
        o0 = wmma_bf16(pa, load_frag(vbase + (size_t)( 0 + l15) * KVL + kv0, hv), o0);
        o1 = wmma_bf16(pa, load_frag(vbase + (size_t)(16 + l15) * KVL + kv0, hv), o1);
        o2 = wmma_bf16(pa, load_frag(vbase + (size_t)(32 + l15) * KVL + kv0, hv), o2);
        o3 = wmma_bf16(pa, load_frag(vbase + (size_t)(48 + l15) * KVL + kv0, hv), o3);
    }

    int b = bh >> 4, h = bh & 15;
    f32x8 os[4] = {o0, o1, o2, o3};
    float inv[8];
    #pragma unroll
    for (int v = 0; v < 8; ++v) inv[v] = __builtin_amdgcn_rcpf(osum[v]);
    #pragma unroll
    for (int t = 0; t < 4; ++t) {
        #pragma unroll
        for (int v = 0; v < 8; ++v) {
            float y = os[t][v] * inv[v];
            int s   = qt * 16 + v + 8 * hv;
            int col = h * HD + t * 16 + l15;
            attnb[((size_t)(b * SS + s)) * DM + col] = (__bf16)y;
        }
    }
}

extern "C" void kernel_launch(void* const* d_in, const int* in_sizes, int n_in,
                              void* d_out, int out_size, void* d_ws, size_t ws_size,
                              hipStream_t stream) {
    const float* x  = (const float*)d_in[0];
    const float* xa = (const float*)d_in[1];
    const float* Wq = (const float*)d_in[2];
    const float* bq = (const float*)d_in[3];
    const float* Wk = (const float*)d_in[4];
    const float* Wv = (const float*)d_in[5];
    const float* bv = (const float*)d_in[6];
    const float* Wo = (const float*)d_in[7];
    const float* bo = (const float*)d_in[8];

    float* out   = (float*)d_out;
    float* k_out = out + (size_t)BB * SS * DM;              // [B,H,KV,HD]
    float* v_out = k_out + (size_t)BB * HH * KVL * HD;      // [B,H,KV,HD]

    char* ws = (char*)d_ws;
    const size_t MB = 1ull << 20;
    __bf16* xb    = (__bf16*)(ws);                 //  4 MB  [B*S, D]
    __bf16* xab   = (__bf16*)(ws +   4 * MB);      // 32 MB  [B*KV, D]
    __bf16* Wqb   = (__bf16*)(ws +  36 * MB);      //  2 MB
    __bf16* Wkb   = (__bf16*)(ws +  38 * MB);      //  2 MB
    __bf16* Wvb   = (__bf16*)(ws +  40 * MB);      //  2 MB
    __bf16* Wob   = (__bf16*)(ws +  42 * MB);      //  2 MB
    __bf16* qbuf  = (__bf16*)(ws +  44 * MB);      //  4 MB  [B,H,S,HD]
    __bf16* kbuf  = (__bf16*)(ws +  48 * MB);      // 32 MB  [B,H,KV,HD]
    __bf16* vTb   = (__bf16*)(ws +  80 * MB);      // 32 MB  [B,H,HD,KV]
    __bf16* attnb = (__bf16*)(ws + 112 * MB);      //  4 MB  [B*S, D]

    // 1) f32 -> bf16 staging
    int n_x  = BB * SS * DM;        // 2,097,152
    int n_xa = BB * KVL * DM;       // 16,777,216
    int n_w  = DM * DM;             // 1,048,576
    cvt_kernel<<<n_x  / 1024, 256, 0, stream>>>(x,  xb,  n_x);
    cvt_kernel<<<n_xa / 1024, 256, 0, stream>>>(xa, xab, n_xa);
    cvt_kernel<<<n_w  / 1024, 256, 0, stream>>>(Wq, Wqb, n_w);
    cvt_kernel<<<n_w  / 1024, 256, 0, stream>>>(Wk, Wkb, n_w);
    cvt_kernel<<<n_w  / 1024, 256, 0, stream>>>(Wv, Wvb, n_w);
    cvt_kernel<<<n_w  / 1024, 256, 0, stream>>>(Wo, Wob, n_w);

    // 2) projections (wave = 16x64 tile; waves = (rows/16)*(D/64))
    int rows_q  = BB * SS;    // 2048  -> 2048 waves  -> 256 blocks
    int rows_kv = BB * KVL;   // 16384 -> 16384 waves -> 2048 blocks
    proj_kernel<0><<<(rows_q  >> 4) * 16 * 32 / 256, 256, 0, stream>>>(
        xb,  Wqb, bq,      nullptr, qbuf, rows_q);
    proj_kernel<1><<<(rows_kv >> 4) * 16 * 32 / 256, 256, 0, stream>>>(
        xab, Wkb, nullptr, k_out,   kbuf, rows_kv);
    proj_kernel<2><<<(rows_kv >> 4) * 16 * 32 / 256, 256, 0, stream>>>(
        xab, Wvb, bv,      v_out,   vTb,  rows_kv);

    // 3) flash attention: B*H*(S/16) = 2048 waves, 4 waves/block
    attn_kernel<<<512, 128, 0, stream>>>(qbuf, kbuf, vTb, attnb);

    // 4) output projection
    proj_kernel<3><<<(rows_q >> 4) * 16 * 32 / 256, 256, 0, stream>>>(
        attnb, Wob, bo, out, nullptr, rows_q);
}
